// LeoSLM_88218628260693
// MI455X (gfx1250) — matrix operand order repaired and checked by reading, not compiled, not run on hardware
//
#include <hip/hip_runtime.h>
#include <stdint.h>

// ---------------------------------------------------------------------------
// LeoSLM forward for MI455X (gfx1250).
// All dense projections via v_wmma_f32_16x16x32_bf16 (wave32 WMMA), f32 acc.
// B operands pre-packed into WMMA K-pair dword layout so GEMM staging is pure
// 128-bit copies -> double-buffered async global->LDS (ASYNCcnt) overlapping
// the WMMA pipeline; one barrier per 64-deep K-slab (16 WMMAs).
// ---------------------------------------------------------------------------

typedef unsigned short bf16_t;
typedef __bf16 v16bf __attribute__((ext_vector_type(16)));
typedef float  v8f   __attribute__((ext_vector_type(8)));
typedef uint32_t v4u __attribute__((ext_vector_type(4)));
typedef int v4i_t __attribute__((ext_vector_type(4)));

#define B_    2
#define T_    2048
#define D_    1024
#define V_    32000
#define L_    4
#define H_    16
#define DH_   64
#define F_    4096
#define CH_   256
#define NMEM_ 8
#define NSAM_ 8
#define NTOOL_ 16

#define INV_SQRT_DH 0.125f
#define INV_SQRT_D  0.03125f

#if defined(__AMDGCN__) && __has_builtin(__builtin_amdgcn_global_load_async_to_lds_b128)
#define ASYNC_LDS 1
#define ASYNC_CP128(gp, lp)                                                   \
  __builtin_amdgcn_global_load_async_to_lds_b128(                             \
      (__attribute__((address_space(1))) v4i_t*)(gp),                         \
      (__attribute__((address_space(3))) v4i_t*)(lp), 0, 0)
#if __has_builtin(__builtin_amdgcn_s_wait_asynccnt)
#define WAIT_ASYNC() __builtin_amdgcn_s_wait_asynccnt(0)
#else
#define WAIT_ASYNC() asm volatile("s_wait_asynccnt 0x0" ::: "memory")
#endif
#endif

static __device__ __forceinline__ bf16_t f2bf(float f) {
  uint32_t u = __float_as_uint(f);
  u += 0x7FFFu + ((u >> 16) & 1u);          // round-to-nearest-even
  return (bf16_t)(u >> 16);
}
static __device__ __forceinline__ uint32_t pack2bf(float lo, float hi) {
  return (uint32_t)f2bf(lo) | ((uint32_t)f2bf(hi) << 16);
}
static __device__ __forceinline__ float sigm(float x) { return 1.f / (1.f + __expf(-x)); }

// ---------------------------------------------------------------------------
// bf16 WMMA GEMM: C[M,N] = A[M,K] * Bp[K,N], f32 out.
//   A : bf16 row-major [M,K]          (K-pairs naturally dword-packed)
//   Bp: dword [K/2][N], dword = (B[2p][n] | B[2p+1][n]<<16)  (pre-packed)
// Block tile 128x128, K-slab 64 (2 WMMA sub-steps), 256 threads = 8 waves
// (2x4), wave tile 64x32 = 4x2 tiles of v_wmma_f32_16x16x32_bf16.
// Double-buffered LDS staging (2 x 32KB).
// ---------------------------------------------------------------------------
#define BM 128
#define BN 128
#define BK 64

union FragU { uint32_t u[8]; v16bf v; };

__global__ __launch_bounds__(256)
void k_gemm_bf16(const bf16_t* __restrict__ A, const uint32_t* __restrict__ Bp,
                 float* __restrict__ C, int M, int N, int K) {
  __shared__ uint32_t lA[2][BM * (BK / 2)];   // [128][32] dwords (K-pair packed)
  __shared__ uint32_t lB[2][(BK / 2) * BN];   // [32][128] dwords (K-pair packed)

  const int tid  = threadIdx.x;
  const int bm   = blockIdx.y * BM;
  const int bn   = blockIdx.x * BN;
  const int wave = tid >> 5;
  const int lane = tid & 31;
  const int half = lane >> 4;
  const int lw   = lane & 15;
  const int wm   = (wave >> 2) * 64;   // 2 wave rows
  const int wn   = (wave & 3) * 32;    // 4 wave cols

  const bool fast = (bm + BM <= M) && (bn + BN <= N);  // block-uniform

  // stage one 128x64 A tile + 64x128 B tile into LDS buffer sb at K offset kk
  auto stage = [&](int sb, int kk) {
    if (fast) {
      // A: 128 rows x 32 dwords = 1024 x b128 ; 4 per thread
#pragma unroll
      for (int j = 0; j < 4; ++j) {
        int item = tid + j * 256;
        int row = item >> 3, grp = item & 7;
        const uint32_t* ga =
            (const uint32_t*)(A + (size_t)(bm + row) * K + kk) + grp * 4;
        uint32_t* la = &lA[sb][row * (BK / 2) + grp * 4];
#ifdef ASYNC_LDS
        ASYNC_CP128(ga, la);
#else
        *(v4u*)la = *(const v4u*)ga;
#endif
      }
      // B: 32 pair-rows x 128 dwords = 1024 x b128 ; 4 per thread
#pragma unroll
      for (int j = 0; j < 4; ++j) {
        int item = tid + j * 256;
        int pr = item >> 5, g = item & 31;
        const uint32_t* gb = Bp + (size_t)(kk / 2 + pr) * N + bn + g * 4;
        uint32_t* lb = &lB[sb][pr * BN + g * 4];
#ifdef ASYNC_LDS
        ASYNC_CP128(gb, lb);
#else
        *(v4u*)lb = *(const v4u*)gb;
#endif
      }
    } else {
      // guarded path (only partial-M tiles, e.g. the NMEM=8 GEMM)
#pragma unroll
      for (int i = 0; i < 16; ++i) {
        int idx = tid + i * 256;          // 4096 A dwords
        int row = idx >> 5, kp = idx & 31;
        int gm = bm + row;
        uint32_t v = 0;
        if (gm < M) v = *(const uint32_t*)(A + (size_t)gm * K + kk + kp * 2);
        lA[sb][row * (BK / 2) + kp] = v;
      }
#pragma unroll
      for (int i = 0; i < 16; ++i) {
        int idx = tid + i * 256;          // 4096 B dwords
        int pr = idx >> 7, n = idx & 127;
        uint32_t v = 0;
        if (bn + n < N) v = Bp[(size_t)(kk / 2 + pr) * N + bn + n];
        lB[sb][pr * BN + n] = v;
      }
    }
  };

  const v8f zero8 = {0.f, 0.f, 0.f, 0.f, 0.f, 0.f, 0.f, 0.f};
  v8f acc[4][2];
#pragma unroll
  for (int mt = 0; mt < 4; ++mt)
#pragma unroll
    for (int nt = 0; nt < 2; ++nt) acc[mt][nt] = zero8;

  // ---- prologue: stage first slab
  stage(0, 0);
#ifdef ASYNC_LDS
  if (fast) WAIT_ASYNC();
#endif
  __syncthreads();

  for (int k0 = 0; k0 < K; k0 += BK) {
    const int p = (k0 / BK) & 1;

    // ---- issue next slab's async copies into the other buffer (overlaps WMMA)
    if (k0 + BK < K) {
      stage(p ^ 1, k0 + BK);
      if (fast && k0 + 2 * BK < K)   // gfx1250 global_prefetch_b8, 2 slabs ahead
        __builtin_prefetch(A + (size_t)(bm + (tid >> 1)) * K + k0 + 2 * BK);
    }

    // ---- two WMMA K=32 sub-steps per slab
#pragma unroll
    for (int s = 0; s < 2; ++s) {
      // fragments in contiguous k order (k = 16*half + element), same for A and B
      FragU bf[2];
#pragma unroll
      for (int nt = 0; nt < 2; ++nt)
#pragma unroll
        for (int v = 0; v < 8; ++v)
          bf[nt].u[v] = lB[p][(s * 16 + half * 8 + v) * BN + wn + nt * 16 + lw];

      FragU af[4];
#pragma unroll
      for (int mt = 0; mt < 4; ++mt)
#pragma unroll
        for (int v = 0; v < 8; ++v) {
          int kv = half * 16 + 2 * v;
          af[mt].u[v] =
              lA[p][(wm + mt * 16 + lw) * (BK / 2) + s * 16 + (kv >> 1)];
        }

#pragma unroll
      for (int mt = 0; mt < 4; ++mt)
#pragma unroll
        for (int nt = 0; nt < 2; ++nt)
          acc[mt][nt] = __builtin_amdgcn_wmma_f32_16x16x32_bf16(
              false, af[mt].v, false, bf[nt].v, (short)0, acc[mt][nt], false,
              false);
    }

    // one barrier per slab: all waves done reading buf p, all async writes
    // to buf p^1 complete (each wave drains its own ASYNCcnt first)
#ifdef ASYNC_LDS
    if (fast) WAIT_ASYNC();
#endif
    __syncthreads();
  }

  // ---- epilogue: C/D layout VGPR r -> M = half*8 + r
  if (fast) {
#pragma unroll
    for (int mt = 0; mt < 4; ++mt)
#pragma unroll
      for (int nt = 0; nt < 2; ++nt) {
        size_t base = (size_t)(bm + wm + mt * 16 + half * 8) * N +
                      (size_t)(bn + wn + nt * 16 + lw);
#pragma unroll
        for (int r = 0; r < 8; ++r)
          C[base + (size_t)r * N] = acc[mt][nt][r];
      }
  } else {
#pragma unroll
    for (int mt = 0; mt < 4; ++mt)
#pragma unroll
      for (int nt = 0; nt < 2; ++nt)
#pragma unroll
        for (int r = 0; r < 8; ++r) {
          int m = bm + wm + mt * 16 + half * 8 + r;
          int n = bn + wn + nt * 16 + lw;
          if (m < M && n < N) C[(size_t)m * N + n] = acc[mt][nt][r];
        }
  }
}

// ---------------------------------------------------------------------------
// Weight packing (once per launch)
// ---------------------------------------------------------------------------
// src f32 [K,N] -> dst dword [K/2][N]
__global__ void k_pack_bf16(const float* __restrict__ s, uint32_t* __restrict__ d,
                            int Kn, int Nn) {
  long long i = (long long)blockIdx.x * blockDim.x + threadIdx.x;
  if (i >= (long long)(Kn / 2) * Nn) return;
  int p = (int)(i / Nn), n = (int)(i % Nn);
  d[i] = pack2bf(s[(size_t)(2 * p) * Nn + n], s[(size_t)(2 * p + 1) * Nn + n]);
}

// transposed pack: src f32 [V,D]; B = src^T (K=D, N=V); dst dword [D/2][V]
__global__ void k_packT_bf16(const float* __restrict__ s, uint32_t* __restrict__ d,
                             int Vv, int Dd) {
  long long i = (long long)blockIdx.x * blockDim.x + threadIdx.x;
  if (i >= (long long)(Dd / 2) * Vv) return;
  int p = (int)(i / Vv), v = (int)(i % Vv);
  const float* row = s + (size_t)v * Dd + 2 * p;
  d[i] = pack2bf(row[0], row[1]);
}

// ---------------------------------------------------------------------------
// Elementwise / small kernels
// ---------------------------------------------------------------------------
__global__ void k_f32_to_bf16(const float* __restrict__ s, bf16_t* __restrict__ d, long long n) {
  long long i = (long long)blockIdx.x * blockDim.x + threadIdx.x;
  if (i < n) d[i] = f2bf(s[i]);
}

__global__ void k_masks(const int* __restrict__ ids, float* __restrict__ think,
                        float* __restrict__ tool) {
  int b = threadIdx.x;
  if (b >= B_) return;
  bool it = false, il = false;
  for (int t = 0; t < T_; ++t) {
    int tok = ids[b * T_ + t];
    it = it || (tok == 31996);
    think[b * T_ + t] = it ? 1.f : 0.f;
    it = it && (tok != 31997);
    il = il || (tok == 31998);
    tool[b * T_ + t] = il ? 1.f : 0.f;
    il = il && (tok != 31999);
  }
}

__global__ void k_embed(const int* __restrict__ ids, const float* __restrict__ emb,
                        float* __restrict__ x, bf16_t* __restrict__ xbf) {
  long long i = (long long)blockIdx.x * blockDim.x + threadIdx.x;
  if (i >= (long long)B_ * T_ * D_) return;
  int row = (int)(i / D_), d = (int)(i % D_);
  float v = emb[(size_t)ids[row] * D_ + d];
  x[i] = v;
  xbf[i] = f2bf(v);
}

__global__ __launch_bounds__(256)
void k_rms_bf16(const float* __restrict__ x, const float* __restrict__ w,
                bf16_t* __restrict__ out, int Dd) {
  __shared__ float red[256];
  int r = blockIdx.x;
  const float* xr = x + (size_t)r * Dd;
  float s = 0.f;
  for (int d = threadIdx.x; d < Dd; d += 256) { float v = xr[d]; s += v * v; }
  red[threadIdx.x] = s; __syncthreads();
  for (int st = 128; st > 0; st >>= 1) {
    if (threadIdx.x < st) red[threadIdx.x] += red[threadIdx.x + st];
    __syncthreads();
  }
  float rn = rsqrtf(red[0] / Dd + 1e-6f);
  for (int d = threadIdx.x; d < Dd; d += 256)
    out[(size_t)r * Dd + d] = f2bf(xr[d] * rn * w[d]);
}

// post-GEMM ect: h = tanh(hpre + b); U = sigmoid(h.w_u (+ tb*think)); optional h out
__global__ __launch_bounds__(256)
void k_ect_post(const float* __restrict__ hpre, const float* __restrict__ bias,
                const float* __restrict__ w_u, const float* __restrict__ tbp,
                const float* __restrict__ think, float* __restrict__ Uout,
                float* __restrict__ hout, int Dd) {
  __shared__ float red[256];
  int r = blockIdx.x;
  const float* hr = hpre + (size_t)r * Dd;
  float s = 0.f;
  for (int d = threadIdx.x; d < Dd; d += 256) {
    float h = tanhf(hr[d] + bias[d]);
    if (hout) hout[(size_t)r * Dd + d] = h;
    s += h * w_u[d];
  }
  red[threadIdx.x] = s; __syncthreads();
  for (int st = 128; st > 0; st >>= 1) {
    if (threadIdx.x < st) red[threadIdx.x] += red[threadIdx.x + st];
    __syncthreads();
  }
  if (threadIdx.x == 0) {
    float ul = red[0];
    if (think) ul += tbp[0] * think[r];
    Uout[r] = sigm(ul);
  }
}

__global__ void k_alpha(const float* __restrict__ U, const float* __restrict__ wap,
                        const float* __restrict__ bap, float* __restrict__ alpha,
                        float* __restrict__ fullAlpha, int sidx, float invL) {
  int i = blockIdx.x * blockDim.x + threadIdx.x;
  if (i >= B_ * CH_) return;
  int b = i / CH_, t = i % CH_;
  float a = sigm(wap[0] * U[(size_t)b * T_ + sidx + t] + bap[0]);
  alpha[i] = a;
  fullAlpha[(size_t)b * T_ + sidx + t] += a * invL;
}

// acc += scale * sum(f(x)); mode 0: x, 1: x^2, 2: (x-0.5)^2
__global__ __launch_bounds__(256)
void k_redadd(const float* __restrict__ x, long long n, int mode, float scale,
              float* __restrict__ acc) {
  __shared__ float red[256];
  float s = 0.f;
  for (long long i = (long long)blockIdx.x * 256 + threadIdx.x; i < n;
       i += (long long)gridDim.x * 256) {
    float v = x[i];
    if (mode == 1) v = v * v;
    else if (mode == 2) { v -= 0.5f; v = v * v; }
    s += v;
  }
  red[threadIdx.x] = s; __syncthreads();
  for (int st = 128; st > 0; st >>= 1) {
    if (threadIdx.x < st) red[threadIdx.x] += red[threadIdx.x + st];
    __syncthreads();
  }
  if (threadIdx.x == 0) atomicAdd(acc, red[0] * scale);
}

// x[b, sidx+t, :] += alpha[b,t] * delta[b,t,:]
__global__ void k_resid(float* __restrict__ x, const float* __restrict__ delta,
                        const float* __restrict__ alpha, int sidx) {
  long long i = (long long)blockIdx.x * blockDim.x + threadIdx.x;
  if (i >= (long long)B_ * CH_ * D_) return;
  int d = (int)(i % D_);
  int row = (int)(i / D_);
  int t = row % CH_, b = row / CH_;
  x[((size_t)b * T_ + sidx + t) * D_ + d] += alpha[b * CH_ + t] * delta[i];
}

__global__ void k_silu_mul(const float* __restrict__ a, const float* __restrict__ b,
                           bf16_t* __restrict__ g, long long n) {
  long long i = (long long)blockIdx.x * blockDim.x + threadIdx.x;
  if (i >= n) return;
  float v = a[i];
  g[i] = f2bf((v * sigm(v)) * b[i]);
}

// per-thread flash attention over (b, h, qi); K <= NMEM_+CH_
__global__ __launch_bounds__(256)
void k_attn(const float* __restrict__ qkv, const float* __restrict__ mqkv, int Mm,
            float* __restrict__ outp) {
  int idx = blockIdx.x * blockDim.x + threadIdx.x;
  if (idx >= B_ * H_ * CH_) return;
  int qi = idx % CH_;
  int h  = (idx / CH_) % H_;
  int b  = idx / (CH_ * H_);

  const float* qrow = qkv + ((size_t)b * CH_ + qi) * 3 * D_ + h * DH_;
  float q[DH_], o[DH_];
#pragma unroll
  for (int d = 0; d < DH_; ++d) { q[d] = qrow[d] * INV_SQRT_DH; o[d] = 0.f; }
  float mx = -1e30f, l = 0.f;
  int kvn = Mm + qi + 1;
  for (int j = 0; j < kvn; ++j) {
    const float* kr;
    if (j < Mm) kr = mqkv + ((size_t)b * NMEM_ + j) * 3 * D_ + D_ + h * DH_;
    else        kr = qkv  + ((size_t)b * CH_ + (j - Mm)) * 3 * D_ + D_ + h * DH_;
    const float* vr = kr + D_;
    float s = 0.f;
#pragma unroll
    for (int d = 0; d < DH_; ++d) s += q[d] * kr[d];
    float nm = fmaxf(mx, s);
    float corr = __expf(mx - nm), e = __expf(s - nm);
    l = l * corr + e;
#pragma unroll
    for (int d = 0; d < DH_; ++d) o[d] = o[d] * corr + e * vr[d];
    mx = nm;
  }
  float inv = 1.f / l;
  float* orow = outp + ((size_t)b * CH_ + qi) * D_ + h * DH_;
#pragma unroll
  for (int d = 0; d < DH_; ++d) orow[d] = o[d] * inv;
}

// softmax attention with a tiny (NM) set of learned queries
__global__ __launch_bounds__(256)
void k_memattn(const float* __restrict__ Q, const float* __restrict__ X,
               const float* __restrict__ Ur, float* __restrict__ memOut,
               int NM, int Tlen, size_t bstrideX, int urStride) {
  __shared__ float p[T_];
  __shared__ float red[256];
  int b = blockIdx.x / NM, m = blockIdx.x % NM;
  const float* Xb = X + (size_t)b * bstrideX;
  const float* q  = Q + (size_t)m * D_;
  for (int t = threadIdx.x; t < Tlen; t += 256) {
    const float* xr = Xb + (size_t)t * D_;
    float s = 0.f;
    for (int d = 0; d < D_; ++d) s += q[d] * xr[d];
    float u = Ur ? Ur[b * urStride + t] : 1.f;
    p[t] = s * u * INV_SQRT_D;
  }
  __syncthreads();
  float mx = -1e30f;
  for (int t = threadIdx.x; t < Tlen; t += 256) mx = fmaxf(mx, p[t]);
  red[threadIdx.x] = mx; __syncthreads();
  for (int st = 128; st > 0; st >>= 1) {
    if (threadIdx.x < st) red[threadIdx.x] = fmaxf(red[threadIdx.x], red[threadIdx.x + st]);
    __syncthreads();
  }
  mx = red[0]; __syncthreads();
  float sum = 0.f;
  for (int t = threadIdx.x; t < Tlen; t += 256) { float e = __expf(p[t] - mx); p[t] = e; sum += e; }
  red[threadIdx.x] = sum; __syncthreads();
  for (int st = 128; st > 0; st >>= 1) {
    if (threadIdx.x < st) red[threadIdx.x] += red[threadIdx.x + st];
    __syncthreads();
  }
  float inv = 1.f / red[0]; __syncthreads();
  for (int d = threadIdx.x; d < D_; d += 256) {
    float a = 0.f;
    for (int t = 0; t < Tlen; ++t) a += p[t] * Xb[(size_t)t * D_ + d];
    memOut[((size_t)b * NM + m) * D_ + d] = a * inv;
  }
}

__global__ __launch_bounds__(256)
void k_gate(const float* __restrict__ hid, const float* __restrict__ ecth,
            const float* __restrict__ wg, const float* __restrict__ Ufin,
            float* __restrict__ gate) {
  __shared__ float red[256];
  int r = blockIdx.x;
  float s = 0.f;
  for (int d = threadIdx.x; d < D_; d += 256)
    s += (hid[(size_t)r * D_ + d] + ecth[(size_t)r * D_ + d]) * wg[d];
  red[threadIdx.x] = s; __syncthreads();
  for (int st = 128; st > 0; st >>= 1) {
    if (threadIdx.x < st) red[threadIdx.x] += red[threadIdx.x + st];
    __syncthreads();
  }
  if (threadIdx.x == 0) gate[r] = sigm(red[0] + Ufin[r]);
}

__global__ void k_tool(const float* __restrict__ hid, const float* __restrict__ Wt,
                       float* __restrict__ out) {
  int i = blockIdx.x * blockDim.x + threadIdx.x;
  if (i >= B_ * T_ * NTOOL_) return;
  int r = i / NTOOL_, j = i % NTOOL_;
  float s = 0.f;
  const float* hr = hid + (size_t)r * D_;
  for (int d = 0; d < D_; ++d) s += hr[d] * Wt[d * NTOOL_ + j];
  out[i] = s;
}

// ---------------------------------------------------------------------------
// Host orchestration
// ---------------------------------------------------------------------------
extern "C" void kernel_launch(void* const* d_in, const int* in_sizes, int n_in,
                              void* d_out, int out_size, void* d_ws, size_t ws_size,
                              hipStream_t stream) {
  (void)in_sizes; (void)n_in; (void)out_size; (void)ws_size;

  const int* ids      = (const int*)d_in[0];
  const float* emb    = (const float*)d_in[1];
  const float* ectW   = (const float*)d_in[2];
  const float* ectb   = (const float*)d_in[3];
  const float* ectwu  = (const float*)d_in[4];
  const float* ecttb  = (const float*)d_in[6];
  auto blk = [&](int l, int j) { return (const float*)d_in[7 + l * 9 + j]; };
  // j: 0 ln1, 1 ln2, 2 Wqkv, 3 Wo, 4 W1, 5 W3, 6 W2, 7 w_alpha, 8 b_alpha
  const float* tdm_q  = (const float*)d_in[43];
  const float* sam_q  = (const float*)d_in[44];
  const float* wgate  = (const float*)d_in[45];
  const float* wtool  = (const float*)d_in[46];
  const float* finalw = (const float*)d_in[47];
  const float* wdiff  = (const float*)d_in[48];

  float* out = (float*)d_out;
  const size_t BT = (size_t)B_ * T_;
  const size_t BTV = BT * V_;
  float* out_ar     = out;
  float* out_diff   = out + BTV;
  float* out_U      = out + 2 * BTV;
  float* out_aux    = out_U + BT;
  float* out_think  = out_aux + 1;
  float* out_tool   = out_think + BT;
  float* out_gate   = out_tool + BT;
  float* out_toollg = out_gate + BT;
  float* out_alpha  = out_toollg + BT * NTOOL_;

  // bump allocator on workspace
  size_t off = 0;
  auto alloc = [&](size_t bytes) -> void* {
    off = (off + 255) & ~(size_t)255;
    void* p = (char*)d_ws + off;
    off += bytes;
    return p;
  };

  // packed-B dword counts
  const size_t szQKV = (size_t)(D_ / 2) * 3 * D_;
  const size_t szWo  = (size_t)(D_ / 2) * D_;
  const size_t szW1  = (size_t)(D_ / 2) * F_;
  const size_t szW2  = (size_t)(F_ / 2) * D_;
  const size_t szVoc = (size_t)(D_ / 2) * V_;

  float*    x      = (float*)alloc(BT * D_ * 4);
  bf16_t*   xbf    = (bf16_t*)alloc(BT * D_ * 2);
  float*    U      = (float*)alloc(BT * 4);
  uint32_t* wqkvP  = (uint32_t*)alloc(L_ * szQKV * 4);
  uint32_t* woP    = (uint32_t*)alloc(L_ * szWo * 4);
  uint32_t* w1P    = (uint32_t*)alloc(L_ * szW1 * 4);
  uint32_t* w3P    = (uint32_t*)alloc(L_ * szW1 * 4);
  uint32_t* w2P    = (uint32_t*)alloc(L_ * szW2 * 4);
  uint32_t* ectWP  = (uint32_t*)alloc(szWo * 4);
  uint32_t* embTP  = (uint32_t*)alloc(szVoc * 4);
  uint32_t* wdifTP = (uint32_t*)alloc(szVoc * 4);
  float*    memBuf = (float*)alloc((size_t)B_ * NMEM_ * D_ * 4);
  bf16_t*   memBf  = (bf16_t*)alloc((size_t)B_ * NMEM_ * D_ * 2);
  float*    memQkv = (float*)alloc((size_t)B_ * NMEM_ * 3 * D_ * 4);
  bf16_t*   hB     = (bf16_t*)alloc((size_t)B_ * CH_ * D_ * 2);
  float*    qkv    = (float*)alloc((size_t)B_ * CH_ * 3 * D_ * 4);
  float*    attnO  = (float*)alloc((size_t)B_ * CH_ * D_ * 4);
  bf16_t*   aoB    = (bf16_t*)alloc((size_t)B_ * CH_ * D_ * 2);
  float*    tmpD   = (float*)alloc((size_t)B_ * CH_ * D_ * 4);
  float*    ff1    = (float*)alloc((size_t)B_ * CH_ * F_ * 4);
  float*    ff3    = (float*)alloc((size_t)B_ * CH_ * F_ * 4);
  bf16_t*   gB     = (bf16_t*)alloc((size_t)B_ * CH_ * F_ * 2);
  bf16_t*   cB     = (bf16_t*)alloc((size_t)B_ * CH_ * D_ * 2);
  float*    hpreC  = (float*)alloc((size_t)B_ * CH_ * D_ * 4);
  float*    UrB    = (float*)alloc((size_t)B_ * CH_ * 4);
  float*    alphaB = (float*)alloc((size_t)B_ * CH_ * 4);
  float*    hpreF  = (float*)alloc(BT * D_ * 4);
  float*    ecthF  = (float*)alloc(BT * D_ * 4);
  bf16_t*   hidB   = (bf16_t*)alloc(BT * D_ * 2);
  bf16_t*   hnB    = (bf16_t*)alloc(BT * D_ * 2);

  auto conv = [&](const float* s, bf16_t* d, size_t n) {
    k_f32_to_bf16<<<(unsigned)((n + 255) / 256), 256, 0, stream>>>(s, d, (long long)n);
  };
  auto pack = [&](const float* s, uint32_t* d, int Kn, int Nn) {
    long long n = (long long)(Kn / 2) * Nn;
    k_pack_bf16<<<(unsigned)((n + 255) / 256), 256, 0, stream>>>(s, d, Kn, Nn);
  };
  auto gemm = [&](const bf16_t* A, const uint32_t* Bp, float* Cm, int M, int N, int K) {
    dim3 g((N + BN - 1) / BN, (M + BM - 1) / BM);
    k_gemm_bf16<<<g, 256, 0, stream>>>(A, Bp, Cm, M, N, K);
  };

  // zero accumulated outputs (harness poisons buffers)
  (void)hipMemsetAsync(out_aux, 0, sizeof(float), stream);
  (void)hipMemsetAsync(out_alpha, 0, BT * sizeof(float), stream);

  // ---- weight packing (f32 -> WMMA K-pair bf16 dwords)
  for (int l = 0; l < L_; ++l) {
    pack(blk(l, 2), wqkvP + (size_t)l * szQKV, D_, 3 * D_);
    pack(blk(l, 3), woP   + (size_t)l * szWo,  D_, D_);
    pack(blk(l, 4), w1P   + (size_t)l * szW1,  D_, F_);
    pack(blk(l, 5), w3P   + (size_t)l * szW1,  D_, F_);
    pack(blk(l, 6), w2P   + (size_t)l * szW2,  F_, D_);
  }
  pack(ectW, ectWP, D_, D_);
  {
    long long n = (long long)(D_ / 2) * V_;
    k_packT_bf16<<<(unsigned)((n + 255) / 256), 256, 0, stream>>>(emb, embTP, V_, D_);
    k_packT_bf16<<<(unsigned)((n + 255) / 256), 256, 0, stream>>>(wdiff, wdifTP, V_, D_);
  }

  // ---- masks, embeddings
  k_masks<<<1, 32, 0, stream>>>(ids, out_think, out_tool);
  {
    long long n = (long long)BT * D_;
    k_embed<<<(unsigned)((n + 255) / 256), 256, 0, stream>>>(ids, emb, x, xbf);
  }

  // ---- SAM prefix memory (on original x) + aux term
  k_memattn<<<B_ * NSAM_, 256, 0, stream>>>(sam_q, x, nullptr, memBuf, NSAM_, T_,
                                            (size_t)T_ * D_, 0);
  k_redadd<<<64, 256, 0, stream>>>(memBuf, (long long)B_ * NSAM_ * D_, 1,
                                   1e-4f / (B_ * NSAM_ * D_), out_aux);

  // ---- initial ect -> U (with think bias)
  gemm(xbf, ectWP, hpreF, (int)BT, D_, D_);
  k_ect_post<<<(unsigned)BT, 256, 0, stream>>>(hpreF, ectb, ectwu, ecttb, out_think,
                                               U, nullptr, D_);

  const size_t CcD = (size_t)CH_ * D_;
  const size_t CcF = (size_t)CH_ * F_;

  // ---- chunk loop (x doubles as the evolving hidden state, in place)
  for (int ci = 0; ci < T_ / CH_; ++ci) {
    int sidx = ci * CH_;
    for (int l = 0; l < L_; ++l) {
      k_alpha<<<(B_ * CH_ + 255) / 256, 256, 0, stream>>>(U, blk(l, 7), blk(l, 8),
                                                          alphaB, out_alpha, sidx,
                                                          1.0f / L_);
      k_redadd<<<64, 256, 0, stream>>>(alphaB, (long long)B_ * CH_, 2,
                                       1.0f / (B_ * CH_), out_aux);
      // rms(chunk, ln1) -> hB (chunk rows contiguous per batch)
      for (int b = 0; b < B_; ++b)
        k_rms_bf16<<<CH_, 256, 0, stream>>>(x + ((size_t)b * T_ + sidx) * D_,
                                            blk(l, 0), hB + b * CcD, D_);
      if (l == 0) {
        k_rms_bf16<<<B_ * NMEM_, 256, 0, stream>>>(memBuf, blk(l, 0), memBf, D_);
        for (int b = 0; b < B_; ++b)
          gemm(memBf + (size_t)b * NMEM_ * D_, wqkvP + (size_t)l * szQKV,
               memQkv + (size_t)b * NMEM_ * 3 * D_, NMEM_, 3 * D_, D_);
      }
      for (int b = 0; b < B_; ++b)
        gemm(hB + b * CcD, wqkvP + (size_t)l * szQKV,
             qkv + (size_t)b * CH_ * 3 * D_, CH_, 3 * D_, D_);
      k_attn<<<(B_ * H_ * CH_ + 255) / 256, 256, 0, stream>>>(
          qkv, (l == 0) ? memQkv : nullptr, (l == 0) ? NMEM_ : 0, attnO);
      conv(attnO, aoB, (size_t)B_ * CcD);
      for (int b = 0; b < B_; ++b)
        gemm(aoB + b * CcD, woP + (size_t)l * szWo, tmpD + b * CcD, CH_, D_, D_);
      {
        long long n = (long long)B_ * CcD;
        k_resid<<<(unsigned)((n + 255) / 256), 256, 0, stream>>>(x, tmpD, alphaB, sidx);
      }
      // FFN
      for (int b = 0; b < B_; ++b)
        k_rms_bf16<<<CH_, 256, 0, stream>>>(x + ((size_t)b * T_ + sidx) * D_,
                                            blk(l, 1), hB + b * CcD, D_);
      for (int b = 0; b < B_; ++b) {
        gemm(hB + b * CcD, w1P + (size_t)l * szW1, ff1 + b * CcF, CH_, F_, D_);
        gemm(hB + b * CcD, w3P + (size_t)l * szW1, ff3 + b * CcF, CH_, F_, D_);
      }
      {
        long long n = (long long)B_ * CcF;
        k_silu_mul<<<(unsigned)((n + 255) / 256), 256, 0, stream>>>(ff1, ff3, gB, n);
      }
      for (int b = 0; b < B_; ++b)
        gemm(gB + b * CcF, w2P + (size_t)l * szW2, tmpD + b * CcD, CH_, D_, F_);
      {
        long long n = (long long)B_ * CcD;
        k_resid<<<(unsigned)((n + 255) / 256), 256, 0, stream>>>(x, tmpD, alphaB, sidx);
      }
    }
    if (ci < T_ / CH_ - 1) {
      // ect(chunk) -> Ur ; TDM memory update ; aux += 0.01*mean(Ur)
      for (int b = 0; b < B_; ++b)
        conv(x + ((size_t)b * T_ + sidx) * D_, cB + b * CcD, CcD);
      for (int b = 0; b < B_; ++b)
        gemm(cB + b * CcD, ectWP, hpreC + b * CcD, CH_, D_, D_);
      k_ect_post<<<B_ * CH_, 256, 0, stream>>>(hpreC, ectb, ectwu, ecttb, nullptr,
                                               UrB, nullptr, D_);
      k_redadd<<<64, 256, 0, stream>>>(UrB, (long long)B_ * CH_, 0,
                                       0.01f / (B_ * CH_), out_aux);
      k_memattn<<<B_ * NMEM_, 256, 0, stream>>>(tdm_q, x + (size_t)sidx * D_, UrB,
                                                memBuf, NMEM_, CH_, (size_t)T_ * D_,
                                                CH_);
    }
  }

  // ---- final heads
  conv(x, hidB, BT * D_);
  gemm(hidB, ectWP, hpreF, (int)BT, D_, D_);
  k_ect_post<<<(unsigned)BT, 256, 0, stream>>>(hpreF, ectb, ectwu, ecttb, out_think,
                                               out_U, ecthF, D_);
  k_gate<<<(unsigned)BT, 256, 0, stream>>>(x, ecthF, wgate, out_U, out_gate);
  k_tool<<<(unsigned)((BT * NTOOL_ + 255) / 256), 256, 0, stream>>>(x, wtool, out_toollg);
  k_rms_bf16<<<(unsigned)BT, 256, 0, stream>>>(x, finalw, hnB, D_);
  gemm(hnB, embTP, out_ar, (int)BT, V_, D_);
  gemm(hnB, wdifTP, out_diff, (int)BT, V_, D_);
}
